// HashGridRadianceField_65661460021930
// MI455X (gfx1250) — matrix operand order repaired and checked by reading, not compiled, not run on hardware
//
#include <hip/hip_runtime.h>

typedef __attribute__((ext_vector_type(16))) _Float16 v16h;
typedef __attribute__((ext_vector_type(8)))  float    v8f;
typedef __attribute__((ext_vector_type(4)))  unsigned v4u;

#define N_LEVELS  16
#define LOG2_T    19
#define TSIZE     (1u << LOG2_T)
#define NEUR      64
#define IN_DIM    32
#define BLOCK     256
#define PPB       256          // points per block
#define WAVES     (BLOCK / 32) // wave32

struct __align__(16) Smem {
  _Float16 feat[PPB][IN_DIM];       // 16 KB  fp16 features (A-matrix source)
  float    sel[PPB];                //  1 KB  inside-AABB selector
  _Float16 W0t[NEUR][IN_DIM];       //  4 KB  transposed [n][k]
  _Float16 W1t[NEUR][NEUR];         //  8 KB
  _Float16 W2t[NEUR][NEUR];         //  8 KB
  _Float16 W3t[16][NEUR];           //  2 KB  output dim padded 4 -> 16
  _Float16 hbuf[WAVES][16][NEUR];   // 16 KB  per-wave activation tile
};                                  // ~55 KB total

__device__ __forceinline__ void lds_fence() {
  // per-wave LDS RAW fence: DS ops are in-order per wave; wait for stores.
  asm volatile("s_wait_dscnt 0" ::: "memory");
}

// A fragment, 16x32 fp16 tile, documented CDNA5 layout:
// lanes 0-15: M=lane, V0-3 K=kOff+0..7, V4-7 K=kOff+16..23
// lanes 16-31: same M, K ranges +8
__device__ __forceinline__ v16h ldsA(const _Float16* base, int ld, int mrow, int hi, int kOff) {
  const _Float16* p = base + mrow * ld + kOff + hi * 8;
  union { v4u u[2]; v16h v; } r;
  r.u[0] = *(const v4u*)(p);        // ds_load_b128
  r.u[1] = *(const v4u*)(p + 16);   // ds_load_b128
  return r.v;
}

// B fragment, 32x16 fp16 tile: lane holds column n=lane%16,
// lanes 0-15 K=0..7,16..23, lanes 16-31 K=8..15,24..31 (matches A layout)
__device__ __forceinline__ v16h ldsB(const _Float16* Wt, int ldk, int n, int hi, int kStep) {
  const _Float16* p = Wt + n * ldk + kStep * 32 + hi * 8;
  union { v4u u[2]; v16h v; } r;
  r.u[0] = *(const v4u*)(p);
  r.u[1] = *(const v4u*)(p + 16);
  return r.v;
}

__device__ __forceinline__ v8f wmma16(v16h a, v16h b, v8f c) {
  return __builtin_amdgcn_wmma_f32_16x16x32_f16(false, a, false, b, (short)0, c, false, false);
}

// C/D layout: VGPR v holds M = v + 8*hi, N = lane%16
__device__ __forceinline__ void storeC_relu(_Float16* hb, int ct, int mrow, int hi, v8f c) {
  #pragma unroll
  for (int v = 0; v < 8; ++v) {
    float x = fmaxf(c[v], 0.f);
    hb[(hi * 8 + v) * NEUR + ct * 16 + mrow] = (_Float16)x;
  }
}

__global__ __launch_bounds__(BLOCK)
void nerf_hash_mlp(const float* __restrict__ positions,
                   const float* __restrict__ table,
                   const float* __restrict__ W0,
                   const float* __restrict__ W1,
                   const float* __restrict__ W2,
                   const float* __restrict__ W3,
                   float* __restrict__ out,
                   int npts)
{
  __shared__ Smem s;
  const int t = threadIdx.x;
  const int blockBase = blockIdx.x * PPB;

  // ---- stage weights into LDS, fp16, transposed [n][k] ----
  for (int i = t; i < IN_DIM * NEUR; i += BLOCK) {
    int k = i / NEUR, n = i % NEUR;
    s.W0t[n][k] = (_Float16)W0[i];
  }
  for (int i = t; i < NEUR * NEUR; i += BLOCK) {
    int k = i / NEUR, n = i % NEUR;
    s.W1t[n][k] = (_Float16)W1[i];
    s.W2t[n][k] = (_Float16)W2[i];
  }
  for (int i = t; i < 16 * NEUR; i += BLOCK) {
    int n = i / NEUR, k = i % NEUR;
    s.W3t[n][k] = (n < 4) ? (_Float16)W3[k * 4 + n] : (_Float16)0.f;
  }

  // ---- hash-grid encode: one point per thread (gathers hit 64MB L2-resident table) ----
  const int gid = blockBase + t;
  const float x = (positions[gid * 3 + 0] + 1.f) * 0.5f;
  const float y = (positions[gid * 3 + 1] + 1.f) * 0.5f;
  const float z = (positions[gid * 3 + 2] + 1.f) * 0.5f;
  s.sel[t] = (x > 0.f && x < 1.f && y > 0.f && y < 1.f && z > 0.f && z < 1.f) ? 1.f : 0.f;

  const float2* __restrict__ tb = (const float2*)table;
  #pragma unroll 2
  for (int l = 0; l < N_LEVELS; ++l) {
    float res = floorf(16.f * exp2f((8.f / 15.f) * (float)l)); // 16 * 256^(l/15)
    float sx = x * res, sy = y * res, sz = z * res;
    float fx = floorf(sx), fy = floorf(sy), fz = floorf(sz);
    float wx = sx - fx, wy = sy - fy, wz = sz - fz;
    unsigned cx = (unsigned)fx, cy = (unsigned)fy, cz = (unsigned)fz;
    unsigned hy0 = cy * 2654435761u, hy1 = (cy + 1u) * 2654435761u;
    unsigned hz0 = cz * 805459861u,  hz1 = (cz + 1u) * 805459861u;
    unsigned base = (unsigned)l << LOG2_T;
    float f0 = 0.f, f1 = 0.f;
    #pragma unroll
    for (int c = 0; c < 8; ++c) {
      unsigned ix  = cx + (unsigned)(c & 1);
      unsigned h   = ix ^ ((c & 2) ? hy1 : hy0) ^ ((c & 4) ? hz1 : hz0);
      unsigned idx = h & (TSIZE - 1u);
      float2 e = tb[base + idx];                  // global_load_b64 gather
      float w = ((c & 1) ? wx : 1.f - wx)
              * ((c & 2) ? wy : 1.f - wy)
              * ((c & 4) ? wz : 1.f - wz);
      f0 += e.x * w;
      f1 += e.y * w;
    }
    s.feat[t][2 * l + 0] = (_Float16)f0;
    s.feat[t][2 * l + 1] = (_Float16)f1;
  }

  __syncthreads();

  // ---- fused MLP via WMMA: each wave owns rows [wave*32, wave*32+32) ----
  const int wave = t >> 5;
  const int lane = t & 31;
  const int mrow = lane & 15;
  const int hi   = lane >> 4;
  _Float16* hb = &s.hbuf[wave][0][0];
  const v8f czero = {};

  for (int tile = 0; tile < 2; ++tile) {
    const int r0 = wave * 32 + tile * 16;

    // layer 0: [16x32] x [32x64], K=32 = one WMMA per 16-col tile
    {
      v16h a = ldsA(&s.feat[r0][0], IN_DIM, mrow, hi, 0);
      #pragma unroll
      for (int ct = 0; ct < 4; ++ct) {
        v16h b = ldsB(&s.W0t[0][0], IN_DIM, ct * 16 + mrow, hi, 0);
        v8f c = czero;
        c = wmma16(a, b, c);
        storeC_relu(hb, ct, mrow, hi, c);
      }
      lds_fence();
    }

    // layers 1,2: [16x64] x [64x64], 2 K-steps x 4 col tiles, in-place in hb
    #pragma unroll
    for (int layer = 0; layer < 2; ++layer) {
      const _Float16* Wt = (layer == 0) ? &s.W1t[0][0] : &s.W2t[0][0];
      v16h a0 = ldsA(hb, NEUR, mrow, hi, 0);
      v16h a1 = ldsA(hb, NEUR, mrow, hi, 32);
      v8f c[4];
      #pragma unroll
      for (int ct = 0; ct < 4; ++ct) {
        v16h b0 = ldsB(Wt, NEUR, ct * 16 + mrow, hi, 0);
        v16h b1 = ldsB(Wt, NEUR, ct * 16 + mrow, hi, 1);
        c[ct] = czero;
        c[ct] = wmma16(a0, b0, c[ct]);
        c[ct] = wmma16(a1, b1, c[ct]);
      }
      #pragma unroll
      for (int ct = 0; ct < 4; ++ct)
        storeC_relu(hb, ct, mrow, hi, c[ct]);
      lds_fence();
    }

    // layer 3: [16x64] x [64x16] (cols 4..15 zero-padded)
    {
      v16h a0 = ldsA(hb, NEUR, mrow, hi, 0);
      v16h a1 = ldsA(hb, NEUR, mrow, hi, 32);
      v16h b0 = ldsB(&s.W3t[0][0], NEUR, mrow, hi, 0);
      v16h b1 = ldsB(&s.W3t[0][0], NEUR, mrow, hi, 1);
      v8f c = czero;
      c = wmma16(a0, b0, c);
      c = wmma16(a1, b1, c);

      // heads: lanes with N<3 -> sigmoid rgb, N==3 -> trunc_exp density * selector
      if (mrow < 4) {
        #pragma unroll
        for (int v = 0; v < 8; ++v) {
          int lr = r0 + hi * 8 + v;
          int g  = blockBase + lr;
          float xv = c[v];
          if (mrow < 3) out[g * 3 + mrow] = 1.f / (1.f + expf(-xv));
          else          out[3 * npts + g] = expf(xv - 1.f) * s.sel[lr];
        }
      }
    }
  }
}

extern "C" void kernel_launch(void* const* d_in, const int* in_sizes, int n_in,
                              void* d_out, int out_size, void* d_ws, size_t ws_size,
                              hipStream_t stream) {
  (void)n_in; (void)d_ws; (void)ws_size; (void)out_size;
  const float* positions = (const float*)d_in[0];
  const float* table     = (const float*)d_in[1];
  const float* W0        = (const float*)d_in[2];
  const float* W1        = (const float*)d_in[3];
  const float* W2        = (const float*)d_in[4];
  const float* W3        = (const float*)d_in[5];
  float* out = (float*)d_out;

  const int npts = in_sizes[0] / 3;           // 524288
  const int nblocks = npts / PPB;             // 2048 full blocks (EXEC all-1s for WMMA)
  nerf_hash_mlp<<<dim3(nblocks), dim3(BLOCK), 0, stream>>>(
      positions, table, W0, W1, W2, W3, out, npts);
}
